// MyRGN_76295799046448
// MI455X (gfx1250) — matrix operand order repaired and checked by reading, not compiled, not run on hardware
//
#include <hip/hip_runtime.h>
#include <hip/hip_bf16.h>
#include <cstdint>

// ---------------------------------------------------------------------------
// MyRGN on MI455X (gfx1250).
//
// All five matmuls in the scan cell are state-independent -> hoist into large
// batched bf16 GEMMs over B*T rows (v_wmma_f32_16x16x32_bf16, wave32), then a
// trivially parallel elementwise recurrence over T.
//
// GEMM: 128x128x32 block tiles, 8 waves, 64x32 per wave (8 wmma tiles).
// LDS double-buffered with compile-time buffer indices (K % 64 == 0 for all
// call sites); A tile staged via GLOBAL_LOAD_ASYNC_TO_LDS_B32 (ASYNCcnt),
// B tile register-transposed into LDS. W3/W4/W5 GEMMs fused (N = 3072,
// per-column activation). Batch processed in 4 chunks of 64.
// ---------------------------------------------------------------------------

#define VOCAB 10000
#define EMB   256
#define SEQ   250
#define BATCH 256
#define UNITS 1024

#define CHUNK_B 64
#define NCHUNK  (BATCH / CHUNK_B)
#define ROWS    (CHUNK_B * SEQ)        // 16000 rows per chunk (125 * 128)

#define BM 128
#define BN 128
#define BK 32
#define LDSW 17                         // dwords per LDS row (16 + 1 pad)

#define ACT_RELU    0
#define ACT_TANH    1
#define ACT_SIGMOID 2
#define ACT_IGP     (-1)                // per-column: sigmoid | tanh | relu

typedef __attribute__((ext_vector_type(16))) __bf16 v16bf;
typedef __attribute__((ext_vector_type(8)))  float  v8f;

union Frag { uint32_t u[8]; v16bf v; };

__device__ __forceinline__ uint16_t f2bf(float f) {
    uint32_t u = __float_as_uint(f);
    u += 0x7fffu + ((u >> 16) & 1u);     // round-to-nearest-even
    return (uint16_t)(u >> 16);
}
__device__ __forceinline__ float bf2f(uint16_t h) {
    return __uint_as_float(((uint32_t)h) << 16);
}
__device__ __forceinline__ float fast_sigmoid(float x) {
    return 1.f / (1.f + __expf(-x));                  // v_exp_f32 path
}
__device__ __forceinline__ float fast_tanh(float x) {
    float ax = fabsf(x);
    float t  = 1.f - 2.f / (__expf(2.f * ax) + 1.f);  // exact limit at |x|->inf
    return copysignf(t, x);
}
__device__ __forceinline__ float apply_act(float x, int a) {
    if (a == ACT_RELU) return x > 0.f ? x : 0.f;
    if (a == ACT_TANH) return fast_tanh(x);
    return fast_sigmoid(x);
}

// ---- CDNA5 async global->LDS (ASYNCcnt), bypasses VGPR round-trip ---------
__device__ __forceinline__ void async_dw_to_lds(uint32_t lds_byte_addr, const void* gsrc) {
    asm volatile("global_load_async_to_lds_b32 %0, %1, off"
                 :: "v"(lds_byte_addr), "v"((unsigned long long)(size_t)gsrc)
                 : "memory");
}
__device__ __forceinline__ void wait_async0() {
    asm volatile("s_wait_asynccnt 0x0" ::: "memory");
}

// --------------------------- f32 -> bf16 packer ----------------------------
__global__ void pack_bf16_kernel(const float* __restrict__ src,
                                 uint32_t* __restrict__ dst, int ndw) {
    int gid = blockIdx.x * blockDim.x + threadIdx.x;
    if (gid >= ndw) return;
    dst[gid] = (uint32_t)f2bf(src[2 * gid]) | ((uint32_t)f2bf(src[2 * gid + 1]) << 16);
}

// ---- pack concatenated [W3 | W4 | W5] -> bf16 [EMB][3*UNITS] --------------
__global__ void pack_cat_kernel(const float* __restrict__ W3,
                                const float* __restrict__ W4,
                                const float* __restrict__ W5,
                                uint32_t* __restrict__ dst) {
    int gid = blockIdx.x * blockDim.x + threadIdx.x;   // dword in [EMB][1536]
    if (gid >= EMB * (3 * UNITS / 2)) return;
    int e  = gid / (3 * UNITS / 2);
    int np = gid % (3 * UNITS / 2);
    int n  = 2 * np;                                   // pairs never straddle 1024
    const float* src = (n < UNITS) ? (W3 + (size_t)e * UNITS + n)
                     : (n < 2 * UNITS) ? (W4 + (size_t)e * UNITS + (n - UNITS))
                     : (W5 + (size_t)e * UNITS + (n - 2 * UNITS));
    dst[gid] = (uint32_t)f2bf(src[0]) | ((uint32_t)f2bf(src[1]) << 16);
}

// ---- concatenated bias [b3 | b4 | 0] --------------------------------------
__global__ void pack_bias_kernel(const float* __restrict__ b3,
                                 const float* __restrict__ b4,
                                 float* __restrict__ bc) {
    int n = blockIdx.x * blockDim.x + threadIdx.x;
    if (n >= 3 * UNITS) return;
    bc[n] = (n < UNITS) ? b3[n] : (n < 2 * UNITS) ? b4[n - UNITS] : 0.f;
}

// --------------------- embedding gather + bf16 convert ---------------------
__global__ void embed_gather_kernel(const int* __restrict__ tokens,
                                    const float* __restrict__ emb,
                                    uint32_t* __restrict__ Xh, int b0) {
    int gid = blockIdx.x * blockDim.x + threadIdx.x;   // one dword = 2 elems
    if (gid >= ROWS * (EMB / 2)) return;
    int row = gid / (EMB / 2);
    int e2  = gid % (EMB / 2);
    int b   = b0 + row / SEQ;
    int t   = row % SEQ;
    int tok = tokens[b * SEQ + t];
    const float* s = emb + (size_t)tok * EMB + 2 * e2;
    Xh[gid] = (uint32_t)f2bf(s[0]) | ((uint32_t)f2bf(s[1]) << 16);
}

// ------------------------------- bf16 GEMM ---------------------------------
// C[M,N] = act(A[M,K] @ B[K,N] + bias). A/B/C bf16 (dword pairs), bias f32
// (nullptr = none). M%128==0, N%128==0, K%64==0. act==ACT_IGP selects
// activation per 1024-column block (sigmoid|tanh|relu).
__global__ __launch_bounds__(256)
void gemm_bf16_act_kernel(const uint32_t* __restrict__ A,
                          const uint32_t* __restrict__ Bw,
                          const float* __restrict__ bias,
                          uint16_t* __restrict__ C,
                          int Mrows, int N, int K, int act) {
    __shared__ uint32_t sA[2][BM * LDSW];   // A tile: [row][kpair], padded
    __shared__ uint32_t sB[2][BN * LDSW];   // B tile transposed: [col][kpair]

    const int tid  = threadIdx.x;
    const int lane = tid & 31;
    const int wv   = tid >> 5;           // 8 waves
    const int wm   = wv & 1;             // 2 waves along M (64 rows each)
    const int wn   = wv >> 1;            // 4 waves along N (32 cols each)
    const int hl   = lane >> 4;          // half-wave select
    const int lr   = lane & 15;

    const int n0 = blockIdx.x * BN;
    const int m0 = blockIdx.y * BM;
    const int ldA = K >> 1;              // dwords per A row
    const int ldB = N >> 1;              // dwords per B row

    v8f acc[4][2];
#pragma unroll
    for (int i = 0; i < 4; ++i)
#pragma unroll
        for (int j = 0; j < 2; ++j)
            acc[i][j] = (v8f){0.f,0.f,0.f,0.f,0.f,0.f,0.f,0.f};

    // async-stage A tile (128 rows x 16 dwords) straight into LDS
    auto stageA = [&](uint32_t* dstA, int k0) {
#pragma unroll
        for (int j = 0; j < 8; ++j) {
            int f = j * 256 + tid;
            int r = f >> 4, dw = f & 15;
            async_dw_to_lds((uint32_t)(size_t)&dstA[r * LDSW + dw],
                            &A[(size_t)(m0 + r) * ldA + (k0 >> 1) + dw]);
        }
    };
    // register-transpose B tile (rows k0..k0+31, cols n0..n0+127) into LDS
    auto stageB = [&](uint32_t* dstB, int k0) {
#pragma unroll
        for (int j = 0; j < 4; ++j) {
            int f = j * 256 + tid;
            int npair = f & 63;          // column pair 0..63
            int kp    = f >> 6;          // k pair 0..15
            uint32_t g0 = Bw[(size_t)(k0 + 2 * kp    ) * ldB + (n0 >> 1) + npair];
            uint32_t g1 = Bw[(size_t)(k0 + 2 * kp + 1) * ldB + (n0 >> 1) + npair];
            dstB[(2 * npair    ) * LDSW + kp] = (g0 & 0xffffu) | (g1 << 16);
            dstB[(2 * npair + 1) * LDSW + kp] = (g0 >> 16) | (g1 & 0xffff0000u);
        }
    };
    // fragments per ISA 16-bit A/B VGPR layouts + 8 WMMAs
    auto compute = [&](const uint32_t* srcA, const uint32_t* srcB) {
        Frag a[4], b[2];
#pragma unroll
        for (int mt = 0; mt < 4; ++mt) {
            int row = wm * 64 + mt * 16 + lr;
#pragma unroll
            for (int v = 0; v < 8; ++v) {
                // low lanes: v0..3 -> K0..7, v4..7 -> K16..23; high lanes +8
                int dw = (v < 4 ? v : v + 4) + hl * 4;
                a[mt].u[v] = srcA[row * LDSW + dw];
            }
        }
#pragma unroll
        for (int nt = 0; nt < 2; ++nt) {
            int col = wn * 32 + nt * 16 + lr;
#pragma unroll
            for (int v = 0; v < 8; ++v)
                b[nt].u[v] = srcB[col * LDSW + (v < 4 ? v : v + 4) + hl * 4];
        }
#pragma unroll
        for (int mt = 0; mt < 4; ++mt)
#pragma unroll
            for (int nt = 0; nt < 2; ++nt)
                acc[mt][nt] = __builtin_amdgcn_wmma_f32_16x16x32_bf16(
                    false, a[mt].v, false, b[nt].v,
                    (short)0, acc[mt][nt], false, false);
    };

    // ---- software pipeline, 2 K-steps per iteration, compile-time buffers ----
    stageA(sA[0], 0);
    stageB(sB[0], 0);
    wait_async0();
    __syncthreads();

    for (int k0 = 0; k0 < K; k0 += 2 * BK) {
        stageA(sA[1], k0 + BK);          // prefetch odd step into buffer 1
        stageB(sB[1], k0 + BK);
        compute(sA[0], sB[0]);
        wait_async0();
        __syncthreads();

        if (k0 + 2 * BK < K) {           // prefetch next even step into buffer 0
            stageA(sA[0], k0 + 2 * BK);
            stageB(sB[0], k0 + 2 * BK);
        }
        compute(sA[1], sB[1]);
        wait_async0();
        __syncthreads();
    }

    // ---- epilogue: bias + activation + bf16 store ----
#pragma unroll
    for (int nt = 0; nt < 2; ++nt) {
        int col = n0 + wn * 32 + nt * 16 + lr;
        float bv = bias ? bias[col] : 0.f;
        int a = (act >= 0) ? act
              : ((col >> 10) == 0 ? ACT_SIGMOID
              : ((col >> 10) == 1 ? ACT_TANH : ACT_RELU));
#pragma unroll
        for (int mt = 0; mt < 4; ++mt) {
#pragma unroll
            for (int v = 0; v < 8; ++v) {
                int row = m0 + wm * 64 + mt * 16 + hl * 8 + v;
                C[(size_t)row * N + col] = f2bf(apply_act(acc[mt][nt][v] + bv, a));
            }
        }
    }
}

// ------------------------- elementwise recurrence --------------------------
// s = relu(P[t] + relu(M[t]*s + I[t]*G[t])), 2 units per thread (dword loads)
__global__ void recurrence_kernel(const uint32_t* __restrict__ Mh,
                                  const uint32_t* __restrict__ IGP,
                                  const float* __restrict__ state0,
                                  float* __restrict__ S, int b0) {
    int gid = blockIdx.x * blockDim.x + threadIdx.x;
    if (gid >= CHUNK_B * (UNITS / 2)) return;
    int bl = gid / (UNITS / 2);
    int up = gid % (UNITS / 2);          // pair of units, coalesced
    size_t sb = (size_t)(b0 + bl) * UNITS + 2 * up;
    float s0 = state0[sb], s1 = state0[sb + 1];
    for (int t = 0; t < SEQ; ++t) {
        size_t r   = (size_t)(bl * SEQ + t);
        uint32_t mw = Mh [r * (UNITS / 2) + up];
        size_t ib  = r * (3 * UNITS / 2) + up;
        uint32_t iw = IGP[ib];
        uint32_t gw = IGP[ib + UNITS / 2];
        uint32_t pw = IGP[ib + UNITS];
        float y0 = bf2f((uint16_t)mw) * s0 + bf2f((uint16_t)iw) * bf2f((uint16_t)gw);
        float y1 = bf2f((uint16_t)(mw >> 16)) * s1
                 + bf2f((uint16_t)(iw >> 16)) * bf2f((uint16_t)(gw >> 16));
        y0 = y0 > 0.f ? y0 : 0.f;  y1 = y1 > 0.f ? y1 : 0.f;
        s0 = bf2f((uint16_t)pw)         + y0;  s0 = s0 > 0.f ? s0 : 0.f;
        s1 = bf2f((uint16_t)(pw >> 16)) + y1;  s1 = s1 > 0.f ? s1 : 0.f;
    }
    S[sb] = s0;  S[sb + 1] = s1;
}

// ------------------------------ output head --------------------------------
__global__ void out_head_kernel(const float* __restrict__ S,
                                const float* __restrict__ Wout,
                                const float* __restrict__ bout,
                                float* __restrict__ out) {
    __shared__ float red[256];
    int b = blockIdx.x, tid = threadIdx.x;
    float p = 0.f;
    for (int u = tid; u < UNITS; u += 256)
        p += S[(size_t)b * UNITS + u] * Wout[u];
    red[tid] = p;
    __syncthreads();
    for (int s = 128; s > 0; s >>= 1) {
        if (tid < s) red[tid] += red[tid + s];
        __syncthreads();
    }
    if (tid == 0) out[b] = fast_sigmoid(red[0] + bout[0]);
}

// ------------------------------- launcher ----------------------------------
extern "C" void kernel_launch(void* const* d_in, const int* in_sizes, int n_in,
                              void* d_out, int out_size, void* d_ws, size_t ws_size,
                              hipStream_t stream) {
    const int*   tokens = (const int*)  d_in[0];
    const float* emb    = (const float*)d_in[1];
    const float* W1     = (const float*)d_in[2];
    const float* b1     = (const float*)d_in[3];
    const float* W2     = (const float*)d_in[4];
    const float* b2     = (const float*)d_in[5];
    const float* W3     = (const float*)d_in[6];
    const float* b3     = (const float*)d_in[7];
    const float* W4     = (const float*)d_in[8];
    const float* b4     = (const float*)d_in[9];
    const float* W5     = (const float*)d_in[10];
    const float* Wout   = (const float*)d_in[11];
    const float* bout   = (const float*)d_in[12];
    const float* state0 = (const float*)d_in[13];
    float* out = (float*)d_out;

    // workspace carve-up (256B aligned); peak ~215 MB
    char*  ws  = (char*)d_ws;
    size_t off = 0;
    auto alloc = [&](size_t bytes) -> void* {
        void* p = ws + off;
        off = (off + bytes + 255) & ~(size_t)255;
        return p;
    };
    uint32_t* W1h = (uint32_t*)alloc((size_t)EMB * 2 * UNITS * 2);     // bf16
    uint32_t* W2h = (uint32_t*)alloc((size_t)2 * UNITS * UNITS * 2);
    uint32_t* Wch = (uint32_t*)alloc((size_t)EMB * 3 * UNITS * 2);     // W3|W4|W5
    float*    bc  = (float*)   alloc((size_t)3 * UNITS * 4);           // b3|b4|0
    uint32_t* Xh  = (uint32_t*)alloc((size_t)ROWS * EMB * 2);
    uint16_t* Rh  = (uint16_t*)alloc((size_t)ROWS * 2 * UNITS * 2);
    uint16_t* Mh  = (uint16_t*)alloc((size_t)ROWS * UNITS * 2);
    uint16_t* IGP = (uint16_t*)alloc((size_t)ROWS * 3 * UNITS * 2);    // I|G|P
    float*    S   = (float*)   alloc((size_t)BATCH * UNITS * 4);

    // 1) pack weights/bias to bf16 (deterministic per call)
    {
        int n1 = EMB * 2 * UNITS / 2;
        pack_bf16_kernel<<<(n1 + 255) / 256, 256, 0, stream>>>(W1, W1h, n1);
        int n2 = 2 * UNITS * UNITS / 2;
        pack_bf16_kernel<<<(n2 + 255) / 256, 256, 0, stream>>>(W2, W2h, n2);
        int n3 = EMB * 3 * UNITS / 2;
        pack_cat_kernel<<<(n3 + 255) / 256, 256, 0, stream>>>(W3, W4, W5, Wch);
        pack_bias_kernel<<<(3 * UNITS + 255) / 256, 256, 0, stream>>>(b3, b4, bc);
    }

    // 2) per-chunk pipeline
    for (int c = 0; c < NCHUNK; ++c) {
        int b0 = c * CHUNK_B;

        int gdw = ROWS * (EMB / 2);
        embed_gather_kernel<<<(gdw + 255) / 256, 256, 0, stream>>>(tokens, emb, Xh, b0);

        dim3 blk(256);
        // R = relu(X @ W1 + b1)                       [ROWS, 2048]
        gemm_bf16_act_kernel<<<dim3(2 * UNITS / BN, ROWS / BM), blk, 0, stream>>>(
            Xh, W1h, b1, Rh, ROWS, 2 * UNITS, EMB, ACT_RELU);
        // M = tanh(R @ W2 + b2)                       [ROWS, 1024]
        gemm_bf16_act_kernel<<<dim3(UNITS / BN, ROWS / BM), blk, 0, stream>>>(
            (const uint32_t*)Rh, W2h, b2, Mh, ROWS, UNITS, 2 * UNITS, ACT_TANH);
        // [I|G|P] = act(X @ [W3|W4|W5] + [b3|b4|0])   [ROWS, 3072]
        gemm_bf16_act_kernel<<<dim3(3 * UNITS / BN, ROWS / BM), blk, 0, stream>>>(
            Xh, Wch, bc, IGP, ROWS, 3 * UNITS, EMB, ACT_IGP);

        recurrence_kernel<<<(CHUNK_B * UNITS / 2 + 255) / 256, 256, 0, stream>>>(
            (const uint32_t*)Mh, (const uint32_t*)IGP, state0, S, b0);
    }

    // 3) output head: sigmoid(S @ Wout + bout)
    out_head_kernel<<<BATCH, 256, 0, stream>>>(S, Wout, bout, out);
    (void)in_sizes; (void)n_in; (void)out_size; (void)ws_size;
}
